// MAMBA_18090402250977
// MI455X (gfx1250) — compile-verified, matrix operand order checked by reading, not run on hardware
//
#include <hip/hip_runtime.h>
#include <hip/hip_bf16.h>

// ---------------------------------------------------------------------------
// SS2D / VMamba block for MI455X (gfx1250, wave32, WMMA f16 16x16x32)
// b=256, 16x16 tokens, d=64, d_state=24, dt_rank=24, K=4 dirs, L=256
// ---------------------------------------------------------------------------

typedef __attribute__((ext_vector_type(16))) _Float16 v16h;
typedef __attribute__((ext_vector_type(8)))  float    v8f;

#define DEVINL __device__ __forceinline__

// K index inside a 16x32 f16 A/B fragment for element e (0..15) of this lane.
// Per CDNA5 ISA 7.12.2: lanes 0-15 hold K {0..7,16..23}, lanes 16-31 hold
// K {8..15,24..31}; element pairs pack 2 f16 per VGPR.
DEVINL int frag_koff(int lane, int e) {
    return ((lane & 16) ? 8 : 0) + e + ((e >> 3) << 3); // e<8 -> e ; e>=8 -> e+8
}

DEVINL v8f wmma_f16(v16h a, v16h b, v8f c) {
    return __builtin_amdgcn_wmma_f32_16x16x32_f16(
        /*neg_a=*/false, a, /*neg_b=*/false, b,
        /*c_mod=*/(short)0, c, /*reuse_a=*/false, /*reuse_b=*/false);
}

// l-permutation for direction k (k&2 -> flip, k&1 -> 16x16 transpose).
// Also happens to be the output scatter position for direction k's scan.
DEVINL int perm_l(int k, int l) {
    int ll = (k & 2) ? (255 - l) : l;
    if (k & 1) ll = ((ll & 15) << 4) | (ll >> 4);
    return ll;
}

DEVINL float silu_f(float v) { return v / (1.0f + __expf(-v)); }

// Async memory->LDS copy of 16 bytes per active lane (CDNA5 ASYNCcnt path).
// GV mode: per-lane 64-bit global address, LDS byte offset in VDST VGPR.
DEVINL void async_g2l_b128(unsigned lds_off, const void* gptr) {
    asm volatile("global_load_async_to_lds_b128 %0, %1, off"
                 :
                 : "v"(lds_off), "v"((unsigned long long)(uintptr_t)gptr)
                 : "memory");
}
DEVINL void wait_asynccnt0() {
    asm volatile("s_wait_asynccnt 0x0" ::: "memory");
}

// ---------------------------------------------------------------------------
// Kernel 0: z = SiLU(x_in @ in_proj_w^T)   (b*256 x 64) x (64 x 64)
// one wave per 16-row M tile; 4 N tiles; K=64 (2 chunks)
// ---------------------------------------------------------------------------
__global__ void k_gemm_z(const float* __restrict__ xin,
                         const float* __restrict__ w,
                         float* __restrict__ z) {
    const int lane = threadIdx.x & 31;
    const int m = lane & 15;
    const long row0 = (long)blockIdx.x * 16;
    const float* A = xin + row0 * 64;

    v16h a0, a1;
#pragma unroll
    for (int e = 0; e < 16; ++e) {
        const int kk = frag_koff(lane, e);
        a0[e] = (_Float16)A[m * 64 + kk];
        a1[e] = (_Float16)A[m * 64 + 32 + kk];
    }
#pragma unroll
    for (int nt = 0; nt < 4; ++nt) {
        const int n = nt * 16 + (lane & 15);
        v16h b0, b1;
#pragma unroll
        for (int e = 0; e < 16; ++e) {
            const int kk = frag_koff(lane, e);
            b0[e] = (_Float16)w[n * 64 + kk];        // B(kk,n) = W[n,kk]  ('ed')
            b1[e] = (_Float16)w[n * 64 + 32 + kk];
        }
        v8f acc = {};
        acc = wmma_f16(a0, b0, acc);
        acc = wmma_f16(a1, b1, acc);
        const int rb = (lane & 16) ? 8 : 0;
#pragma unroll
        for (int i = 0; i < 8; ++i)
            z[(row0 + rb + i) * 64 + n] = silu_f(acc[i]);
    }
}

// ---------------------------------------------------------------------------
// Kernel 1: rearrange (b,16,16,64)->(b,4,64,64) + depthwise 3x3 (pad 1) + SiLU
// ---------------------------------------------------------------------------
__global__ void k_conv_dw(const float* __restrict__ xin,
                          const float* __restrict__ cw,
                          const float* __restrict__ cb,
                          float* __restrict__ conv) {
    const int idx = blockIdx.x * blockDim.x + threadIdx.x; // b*4*64*64
    const int w = idx & 63;
    const int h = (idx >> 6) & 63;
    const int c = (idx >> 12) & 3;
    const int b = idx >> 14;
    float s = 0.f;
#pragma unroll
    for (int dy = -1; dy <= 1; ++dy) {
#pragma unroll
        for (int dx = -1; dx <= 1; ++dx) {
            const int hh = h + dy, ww = w + dx;
            if (hh >= 0 && hh < 64 && ww >= 0 && ww < 64) {
                // image[b,c,hh,ww] = x_in[b, hh/4, ww/4, c*16 + (hh%4)*4 + (ww%4)]
                const float v = xin[(((b << 4) + (hh >> 2)) * 16 + (ww >> 2)) * 64 +
                                    (c << 4) + ((hh & 3) << 2) + (ww & 3)];
                s += v * cw[c * 9 + (dy + 1) * 3 + (dx + 1)];
            }
        }
    }
    s += cb[c];
    conv[idx] = silu_f(s);
}

// ---------------------------------------------------------------------------
// Kernel 2: patch embed 4x4 stride-4 (im2col GEMM, K=64) + BN -> xbuf (b,d,l)
// ---------------------------------------------------------------------------
__global__ void k_patch_bn(const float* __restrict__ conv,
                           const float* __restrict__ pw,
                           const float* __restrict__ pb,
                           const float* __restrict__ g,
                           const float* __restrict__ be,
                           const float* __restrict__ mean,
                           const float* __restrict__ var,
                           float* __restrict__ xbuf) {
    const int lane = threadIdx.x & 31;
    const long row0 = (long)blockIdx.x * 16; // over b*256 pixels
    const int b = (int)(row0 >> 8);
    const int lb = (int)(row0 & 255);
    const int m = lane & 15;
    const int lpix = lb + m;
    const int p = lpix >> 4, q = lpix & 15;

    v16h a0, a1;
#pragma unroll
    for (int e = 0; e < 16; ++e) {
        int kk = frag_koff(lane, e);
        int c = kk >> 4, dy = (kk >> 2) & 3, dx = kk & 3;
        a0[e] = (_Float16)conv[(((b * 4 + c) * 64) + p * 4 + dy) * 64 + q * 4 + dx];
        kk += 32;
        c = kk >> 4; dy = (kk >> 2) & 3; dx = kk & 3;
        a1[e] = (_Float16)conv[(((b * 4 + c) * 64) + p * 4 + dy) * 64 + q * 4 + dx];
    }
#pragma unroll
    for (int nt = 0; nt < 4; ++nt) {
        const int n = nt * 16 + (lane & 15);
        v16h b0, b1;
#pragma unroll
        for (int e = 0; e < 16; ++e) {
            const int kk = frag_koff(lane, e);
            b0[e] = (_Float16)pw[n * 64 + kk];
            b1[e] = (_Float16)pw[n * 64 + 32 + kk];
        }
        v8f acc = {};
        acc = wmma_f16(a0, b0, acc);
        acc = wmma_f16(a1, b1, acc);
        const float sc = g[n] * rsqrtf(var[n] + 1e-5f);
        const float sh = be[n] + (pb[n] - mean[n]) * sc;
        const int rb = (lane & 16) ? 8 : 0;
#pragma unroll
        for (int i = 0; i < 8; ++i)
            xbuf[(b * 64 + n) * 256 + lb + rb + i] = acc[i] * sc + sh;
    }
}

// ---------------------------------------------------------------------------
// Kernel 3: per (b,k, 16-l tile): x_dbl = xs @ x_proj^T (72 wide, padded 80),
// split into dts/B/C; then delta = dts @ dt_proj^T (K=24 padded to 32).
// Bs/Cs stored (b,k,l,24); delta stored (b,k,l,64) for coalesced scan reads.
// ---------------------------------------------------------------------------
__global__ void k_xdbl(const float* __restrict__ xbuf,
                       const float* __restrict__ xpw,
                       const float* __restrict__ dtw,
                       float* __restrict__ Bs,
                       float* __restrict__ Cs,
                       float* __restrict__ delta) {
    __shared__ float tmp[16][80];
    const int lane = threadIdx.x & 31;
    const int tile = blockIdx.x & 15;
    const int bk = blockIdx.x >> 4;   // b*4 + k
    const int k = bk & 3;
    const int b = bk >> 2;
    const int l0 = tile * 16;

    const int pl = perm_l(k, l0 + (lane & 15));
    v16h a0, a1;
#pragma unroll
    for (int e = 0; e < 16; ++e) {
        const int kk = frag_koff(lane, e);
        a0[e] = (_Float16)xbuf[(b * 64 + kk) * 256 + pl];
        a1[e] = (_Float16)xbuf[(b * 64 + 32 + kk) * 256 + pl];
    }
    const float* W = xpw + k * 72 * 64;
#pragma unroll
    for (int nt = 0; nt < 5; ++nt) {
        const int c = nt * 16 + (lane & 15);
        v16h b0, b1;
#pragma unroll
        for (int e = 0; e < 16; ++e) {
            const int kk = frag_koff(lane, e);
            b0[e] = (_Float16)((c < 72) ? W[c * 64 + kk] : 0.f);
            b1[e] = (_Float16)((c < 72) ? W[c * 64 + 32 + kk] : 0.f);
        }
        v8f acc = {};
        acc = wmma_f16(a0, b0, acc);
        acc = wmma_f16(a1, b1, acc);
        const int rb = (lane & 16) ? 8 : 0;
#pragma unroll
        for (int i = 0; i < 8; ++i) tmp[rb + i][c] = acc[i];
    }
    __syncthreads();

    const long rowbase = (long)bk * 256 + l0;
    for (int idx = lane; idx < 16 * 24; idx += 32) {
        const int r = idx / 24, c = idx % 24;
        Bs[(rowbase + r) * 24 + c] = tmp[r][24 + c];
        Cs[(rowbase + r) * 24 + c] = tmp[r][48 + c];
    }

    // delta = dts(16x24) @ dtW^T(24x64); single K chunk of 32 (padded)
    v16h a2;
    const int m2 = lane & 15;
#pragma unroll
    for (int e = 0; e < 16; ++e) {
        const int kk = frag_koff(lane, e);
        a2[e] = (_Float16)((kk < 24) ? tmp[m2][kk] : 0.f);
    }
    const float* DW = dtw + k * 64 * 24;
#pragma unroll
    for (int nt = 0; nt < 4; ++nt) {
        const int d = nt * 16 + (lane & 15);
        v16h bb;
#pragma unroll
        for (int e = 0; e < 16; ++e) {
            const int kk = frag_koff(lane, e);
            bb[e] = (_Float16)((kk < 24) ? DW[d * 24 + kk] : 0.f);
        }
        v8f acc = {};
        acc = wmma_f16(a2, bb, acc);
        const int rb = (lane & 16) ? 8 : 0;
#pragma unroll
        for (int i = 0; i < 8; ++i)
            delta[(rowbase + rb + i) * 64 + d] = acc[i];
    }
}

// ---------------------------------------------------------------------------
// Kernel 4: selective scan for one direction k. 64 threads = 64 channels per
// block, one block per batch. h[24] lives in registers; B/C rows for each
// 16-step chunk are staged into LDS with GLOBAL_LOAD_ASYNC_TO_LDS_B128
// (ASYNCcnt path, no VGPR round-trip), completed before each chunk's scan starts.
// Output scattered with perm_l(k,.) (bijection -> race-free). Launched 4x
// sequentially (k=0 writes, k=1..3 accumulate) for determinism.
// ---------------------------------------------------------------------------
__global__ void k_scan(const float* __restrict__ xbuf,
                       const float* __restrict__ Bs,
                       const float* __restrict__ Cs,
                       const float* __restrict__ delta,
                       const float* __restrict__ A_logs,
                       const float* __restrict__ Ds,
                       const float* __restrict__ dbias,
                       float* __restrict__ y,
                       int k, int accFlag) {
    __shared__ __align__(16) float sB[16][24];
    __shared__ __align__(16) float sC[16][24];
    const int d = threadIdx.x;       // 0..63
    const int b = blockIdx.x;
    const int bk = b * 4 + k;

    float Arow[24], h[24];
#pragma unroll
    for (int n = 0; n < 24; ++n) {
        Arow[n] = -__expf(A_logs[(k * 64 + d) * 24 + n]);
        h[n] = 0.f;
    }
    const float dbv = dbias[k * 64 + d];
    const float Dv = Ds[k * 64 + d];
    const float* xrow = xbuf + (b * 64 + d) * 256;
    const long rb = (long)bk * 256;
    const long ybase = (long)(b * 64 + d) * 256;

    const unsigned ldsB = (unsigned)(uintptr_t)&sB[0][0];
    const unsigned ldsC = (unsigned)(uintptr_t)&sC[0][0];

    for (int l0 = 0; l0 < 256; l0 += 16) {
        __syncthreads();   // previous chunk fully consumed before overwrite
        {
            // 16 rows x 24 floats = 1536 B per array = 96 x b128 transfers.
            const float* gB = Bs + (rb + l0) * 24;
            const float* gC = Cs + (rb + l0) * 24;
            for (int i = threadIdx.x; i < 96; i += 64) {
                async_g2l_b128(ldsB + (unsigned)i * 16u, gB + i * 4);
                async_g2l_b128(ldsC + (unsigned)i * 16u, gC + i * 4);
            }
            wait_asynccnt0();
        }
        __syncthreads();
        if (l0 + 16 < 256)  // emits global_prefetch_b8 for next delta chunk
            __builtin_prefetch(&delta[(rb + l0 + 16) * 64 + d], 0, 0);
#pragma unroll 4
        for (int r = 0; r < 16; ++r) {
            const int l = l0 + r;
            const float draw = delta[(rb + l) * 64 + d] + dbv;
            const float dl = (draw > 20.f) ? draw : __logf(1.f + __expf(draw));
            const float u = xrow[perm_l(k, l)];
            const float du = dl * u;
            float yv = 0.f;
#pragma unroll
            for (int n = 0; n < 24; ++n) {
                const float dA = __expf(dl * Arow[n]);
                h[n] = dA * h[n] + du * sB[r][n];
                yv += h[n] * sC[r][n];
            }
            yv += Dv * u;
            const long oidx = ybase + perm_l(k, l);
            if (accFlag) y[oidx] += yv;
            else         y[oidx] = yv;
        }
    }
}

// ---------------------------------------------------------------------------
// Kernel 5: LayerNorm(d=64) + gate with z + out_proj GEMM -> d_out
// ---------------------------------------------------------------------------
__global__ void k_final(const float* __restrict__ y,
                        const float* __restrict__ z,
                        const float* __restrict__ lg,
                        const float* __restrict__ lb,
                        const float* __restrict__ wout,
                        float* __restrict__ out) {
    __shared__ float sy[16][64];
    const int lane = threadIdx.x & 31;
    const long row0 = (long)blockIdx.x * 16;
    const int b = (int)(row0 >> 8);
    const int lbase = (int)(row0 & 255);

    for (int idx = lane; idx < 16 * 64; idx += 32) {
        const int r = idx >> 6, e = idx & 63;
        sy[r][e] = y[(b * 64 + e) * 256 + lbase + r];
    }
    __syncthreads();
    if (lane < 16) {
        const int r = lane;
        float mu = 0.f;
        for (int e = 0; e < 64; ++e) mu += sy[r][e];
        mu *= (1.f / 64.f);
        float var = 0.f;
        for (int e = 0; e < 64; ++e) { const float t = sy[r][e] - mu; var += t * t; }
        var *= (1.f / 64.f);
        const float rs = rsqrtf(var + 1e-5f);
        const float* zr = z + (row0 + r) * 64;
        for (int e = 0; e < 64; ++e)
            sy[r][e] = ((sy[r][e] - mu) * rs * lg[e] + lb[e]) * zr[e];
    }
    __syncthreads();

    const int m = lane & 15;
    v16h a0, a1;
#pragma unroll
    for (int e = 0; e < 16; ++e) {
        const int kk = frag_koff(lane, e);
        a0[e] = (_Float16)sy[m][kk];
        a1[e] = (_Float16)sy[m][32 + kk];
    }
#pragma unroll
    for (int nt = 0; nt < 4; ++nt) {
        const int n = nt * 16 + (lane & 15);
        v16h b0, b1;
#pragma unroll
        for (int e = 0; e < 16; ++e) {
            const int kk = frag_koff(lane, e);
            b0[e] = (_Float16)wout[n * 64 + kk];      // out[.,d]=sum_e y*W[d,e]
            b1[e] = (_Float16)wout[n * 64 + 32 + kk];
        }
        v8f acc = {};
        acc = wmma_f16(a0, b0, acc);
        acc = wmma_f16(a1, b1, acc);
        const int rb = (lane & 16) ? 8 : 0;
#pragma unroll
        for (int i = 0; i < 8; ++i)
            out[(row0 + rb + i) * 64 + n] = acc[i];
    }
}

// ---------------------------------------------------------------------------
extern "C" void kernel_launch(void* const* d_in, const int* in_sizes, int n_in,
                              void* d_out, int out_size, void* d_ws, size_t ws_size,
                              hipStream_t stream) {
    const float* x_in      = (const float*)d_in[0];
    const float* in_proj_w = (const float*)d_in[1];
    const float* conv_w    = (const float*)d_in[2];
    const float* conv_b    = (const float*)d_in[3];
    const float* patch_w   = (const float*)d_in[4];
    const float* patch_b   = (const float*)d_in[5];
    const float* bn_gamma  = (const float*)d_in[6];
    const float* bn_beta   = (const float*)d_in[7];
    const float* bn_mean   = (const float*)d_in[8];
    const float* bn_var    = (const float*)d_in[9];
    const float* x_proj_w  = (const float*)d_in[10];
    const float* dt_proj_w = (const float*)d_in[11];
    const float* dt_proj_b = (const float*)d_in[12];
    const float* A_logs    = (const float*)d_in[13];
    const float* Ds        = (const float*)d_in[14];
    const float* ln_gamma  = (const float*)d_in[15];
    const float* ln_beta   = (const float*)d_in[16];
    const float* out_proj  = (const float*)d_in[17];
    float* out = (float*)d_out;

    // Workspace layout (floats). Total ~46.1M floats (~185 MB) -> fits L2.
    float* ws    = (float*)d_ws;
    float* zbuf  = ws;                      // b*256*64      = 4,194,304
    float* xbuf  = zbuf  + 4194304;         // b*64*256      = 4,194,304
    float* convb = xbuf  + 4194304;         // b*4*64*64     = 4,194,304
    float* Bsb   = convb + 4194304;         // b*4*256*24    = 6,291,456
    float* Csb   = Bsb   + 6291456;         // b*4*256*24    = 6,291,456
    float* dltb  = Csb   + 6291456;         // b*4*256*64    = 16,777,216
    float* ycb   = dltb  + 16777216;        // b*64*256      = 4,194,304

    k_gemm_z<<<4096, 32, 0, stream>>>(x_in, in_proj_w, zbuf);
    k_conv_dw<<<16384, 256, 0, stream>>>(x_in, conv_w, conv_b, convb);
    k_patch_bn<<<4096, 32, 0, stream>>>(convb, patch_w, patch_b,
                                        bn_gamma, bn_beta, bn_mean, bn_var, xbuf);
    k_xdbl<<<16384, 32, 0, stream>>>(xbuf, x_proj_w, dt_proj_w, Bsb, Csb, dltb);
    for (int k = 0; k < 4; ++k)
        k_scan<<<256, 64, 0, stream>>>(xbuf, Bsb, Csb, dltb, A_logs, Ds, dt_proj_b,
                                       ycb, k, k == 0 ? 0 : 1);
    k_final<<<4096, 32, 0, stream>>>(ycb, zbuf, ln_gamma, ln_beta, out_proj, out);
}
